// HeteroGraph_36627481101209
// MI455X (gfx1250) — matrix operand order repaired and checked by reading, not compiled, not run on hardware
//
#include <hip/hip_runtime.h>
#include <hip/hip_bf16.h>
#include <stdint.h>

// ---------------------------------------------------------------------------
// HeteroGraph SAGE forward, MI455X (gfx1250, wave32, WMMA).
//
//  * agg_as / agg_us / deg_ss are layer-invariant -> computed once.
//  * per-layer node math fused into ONE K=256 GEMM:
//      h = relu([agg_ss' | agg_as | agg_us | sub] @ Bcat + bsum)
//    with Bcat = [Wl0^T ; Wl1^T ; Wl2^T ; (Wr0+Wr1+Wr2)^T], pre-swizzled
//    into v_wmma_f32_16x16x32_f16 B-fragment order (f16), bias pre-summed.
//  * All aggregation buffers (51MB) are L2-resident (192MB L2) so the
//    scatter-adds run at L2 atomic speed, and GEMM A-reads hit L2.
//  * A-staging uses v_cvt_pk_rtz_f16_f32 (packed convert) to build LDS
//    fragments with no bit-shuffle tail.
// ---------------------------------------------------------------------------

typedef __attribute__((ext_vector_type(16))) _Float16 v16h;
typedef __attribute__((ext_vector_type(2)))  __fp16   fp16x2;
typedef __attribute__((ext_vector_type(8)))  float    v8f;

#define H 64
#define OUTC 6
#define NRB_KT 8            // K=256 in 8 chunks of 32
#define GEMM_THREADS 128    // 4 wave32 -> 4 column tiles of 16

// packed f32x2 -> f16x2 in one v_cvt_pk_rtz_f16_f32
static __device__ __forceinline__ uint32_t pk2h(float x, float y) {
    union { fp16x2 h; uint32_t u; } p;
    p.h = __builtin_amdgcn_cvt_pkrtz(x, y);
    return p.u;
}

// ---------------------------------------------------------------- zero -----
__global__ void k_zero(float* __restrict__ p, int n) {
    int i = blockIdx.x * blockDim.x + threadIdx.x;
    int stride = gridDim.x * blockDim.x;
    for (; i < n; i += stride) p[i] = 0.0f;
}

// --------------------------------------------------- edge scatter (sum) ----
// one thread per (edge, 4-feature group); float4 gather + 4 f32 atomics.
__global__ void k_scatter(const float4* __restrict__ xsrc,
                          const int* __restrict__ src,
                          const int* __restrict__ dst,
                          float* __restrict__ agg, int E) {
    int tid = blockIdx.x * blockDim.x + threadIdx.x;
    if (tid >= E * 16) return;
    int e = tid >> 4, g = tid & 15;
    float4 v = xsrc[src[e] * 16 + g];
    float* o = agg + dst[e] * H + g * 4;
    atomicAdd(o + 0, v.x);
    atomicAdd(o + 1, v.y);
    atomicAdd(o + 2, v.z);
    atomicAdd(o + 3, v.w);
}

__global__ void k_deg(const int* __restrict__ dst, float* __restrict__ deg, int E) {
    int e = blockIdx.x * blockDim.x + threadIdx.x;
    if (e < E) atomicAdd(deg + dst[e], 1.0f);
}

// ----------------------------------------- weight prep (fragment order) ----
// Bswz layout (uint32 = packed 2x f16):
//   dword index = (((layer*4 + ctile)*8 + ktile)*32 + lane)*8 + v
//   lane: n = ctile*16 + (lane&15), k = ktile*32 + (lane>>4)*16 + 2v (+0,+1)
// a contiguous-k B (32x16) order that the A fragments share, so the GEMM kernel
// loads each fragment as 8 consecutive dwords per lane (2x b128).
__global__ void k_prep_b(const float* __restrict__ Wl,  // [3,3,64,64]
                         const float* __restrict__ Wr,  // [3,3,64,64]
                         uint32_t* __restrict__ Bswz) {
    int tid = blockIdx.x * blockDim.x + threadIdx.x;   // 3*4*8*32*8 = 24576
    if (tid >= 3 * 4 * 8 * 32 * 8) return;
    int v     =  tid        & 7;
    int lane  = (tid >> 3)  & 31;
    int kt    = (tid >> 8)  & 7;
    int c     = (tid >> 11) & 3;
    int layer =  tid >> 13;
    int n  = c * 16 + (lane & 15);
    int k0 = kt * 32 + (lane >> 4) * 16 + 2 * v;
    float val[2];
    #pragma unroll
    for (int t = 0; t < 2; ++t) {
        int k = k0 + t, seg = k >> 6, kk = k & 63;
        float x;
        if (seg < 3) {
            x = Wl[((layer * 3 + seg) * H + n) * H + kk];
        } else {
            x = Wr[((layer * 3 + 0) * H + n) * H + kk]
              + Wr[((layer * 3 + 1) * H + n) * H + kk]
              + Wr[((layer * 3 + 2) * H + n) * H + kk];
        }
        val[t] = x;
    }
    Bswz[tid] = pk2h(val[0], val[1]);
}

__global__ void k_prep_bias(const float* __restrict__ bl,  // [3,3,64]
                            float* __restrict__ bsum) {    // [3,64]
    int tid = blockIdx.x * blockDim.x + threadIdx.x;       // 192
    if (tid >= 3 * H) return;
    int i = tid / H, j = tid % H;
    bsum[tid] = bl[i * 3 * H + 0 * H + j] + bl[i * 3 * H + 1 * H + j]
              + bl[i * 3 * H + 2 * H + j];
}

// --------------------------------------------------- fused WMMA GEMM -------
// out[N,64] = relu([agg_ss' | agg_as | agg_us | cur][N,256] @ B + bias)
// Block = 128 threads (4 waves). Grid-stride over 16-row blocks; each wave
// owns one 16-column tile and keeps its 8 B fragments (64 VGPRs) hoisted.
__global__ void __launch_bounds__(GEMM_THREADS)
k_gemm(const float* __restrict__ aggss, const float* __restrict__ aggas,
       const float* __restrict__ aggus, const float* __restrict__ cur,
       const float* __restrict__ deg,
       const uint32_t* __restrict__ Bl,   // this layer's Bswz (8192 dwords)
       const float* __restrict__ bsum,    // this layer's bias (64)
       float* __restrict__ out, int nrb, int useMean) {
    __shared__ uint32_t sA[16 * 128];      // 16 rows x 256 k as packed f16 pairs

    const int wave = threadIdx.x >> 5;
    const int lane = threadIdx.x & 31;
    const int m    = lane & 15;            // A row / D column-within-tile
    const int hb   = lane >> 4;            // K half selector

    // hoist this wave's 8 B fragments into VGPRs (reused for every row block)
    union Frag { v16h h; uint32_t u[8]; };
    Frag B[NRB_KT];
    {
        const uint32_t* bp = Bl + (wave * 8) * 32 * 8;
        #pragma unroll
        for (int kt = 0; kt < NRB_KT; ++kt)
            #pragma unroll
            for (int v = 0; v < 8; ++v)
                B[kt].u[v] = bp[(kt * 32 + lane) * 8 + v];
    }
    const float bias = bsum[wave * 16 + m];

    for (int rb = blockIdx.x; rb < nrb; rb += gridDim.x) {
        // prefetch next tile's A rows while this tile computes
        int rbn = rb + gridDim.x;
        if (rbn < nrb && threadIdx.x < 4) {
            const float* srcs[4] = { aggss, aggas, aggus, cur };
            __builtin_prefetch(srcs[threadIdx.x] + rbn * 16 * H, 0, 0);
        }

        // ---- stage A tile to LDS: 16 rows x 256 k, f32 -> packed f16 ----
        #pragma unroll
        for (int it = 0; it < 8; ++it) {
            int idx = threadIdx.x + it * GEMM_THREADS;   // 0..1023 float4 slots
            int mr  = idx >> 6;                          // local row 0..15
            int k0  = (idx & 63) * 4;                    // k 0..252
            int seg = k0 >> 6, kk = k0 & 63;
            int row = rb * 16 + mr;
            const float4* sp = (seg == 0) ? (const float4*)aggss
                             : (seg == 1) ? (const float4*)aggas
                             : (seg == 2) ? (const float4*)aggus
                                          : (const float4*)cur;
            float4 x = sp[row * 16 + (kk >> 2)];
            if (seg == 0 && useMean) {
                float s = 1.0f / fmaxf(deg[row], 1.0f);
                x.x *= s; x.y *= s; x.z *= s; x.w *= s;
            }
            sA[mr * 128 + (k0 >> 1) + 0] = pk2h(x.x, x.y);
            sA[mr * 128 + (k0 >> 1) + 1] = pk2h(x.z, x.w);
        }
        __syncthreads();

        // ---- 8x v_wmma_f32_16x16x32_f16, K = 256 ----
        v8f acc;
        #pragma unroll
        for (int r = 0; r < 8; ++r) acc[r] = bias;
        #pragma unroll
        for (int kt = 0; kt < NRB_KT; ++kt) {
            Frag A;
            int base = m * 128 + kt * 16 + hb * 8;
            #pragma unroll
            for (int v = 0; v < 4; ++v) {
                A.u[v]     = sA[base + v];      // K = kt*32 + hb*16 + 0..7
                A.u[4 + v] = sA[base + 4 + v];  // K = kt*32 + hb*16 + 8..15
            }
            acc = __builtin_amdgcn_wmma_f32_16x16x32_f16(
                /*neg_a=*/false, A.h, /*neg_b=*/false, B[kt].h,
                /*c_mod=*/(short)0, acc, /*reuse_a=*/false, /*reuse_b=*/false);
        }

        // ---- ReLU + store D (rows rb*16+M, cols wave*16 + m) ----
        #pragma unroll
        for (int r = 0; r < 8; ++r) {
            int mm = r + hb * 8;
            out[(rb * 16 + mm) * H + wave * 16 + m] = fmaxf(acc[r], 0.0f);
        }
        __syncthreads();
    }
}

// ----------------------------------------------------- classifier head -----
__global__ void k_head(const float* __restrict__ sub,
                       const float* __restrict__ Wf,   // [6,64]
                       const float* __restrict__ bf,   // [6]
                       float* __restrict__ out, int N) {
    __shared__ float sW[OUTC * H];
    __shared__ float sB[OUTC];
    for (int i = threadIdx.x; i < OUTC * H; i += blockDim.x) sW[i] = Wf[i];
    if (threadIdx.x < OUTC) sB[threadIdx.x] = bf[threadIdx.x];
    __syncthreads();

    int node = blockIdx.x * blockDim.x + threadIdx.x;
    int stride = gridDim.x * blockDim.x;
    for (; node < N; node += stride) {
        const float4* xr = (const float4*)(sub + node * H);
        float a[OUTC];
        #pragma unroll
        for (int j = 0; j < OUTC; ++j) a[j] = sB[j];
        #pragma unroll
        for (int q = 0; q < 16; ++q) {
            float4 v = xr[q];
            #pragma unroll
            for (int j = 0; j < OUTC; ++j) {
                const float4* w = (const float4*)(sW + j * H);
                float4 u = w[q];
                a[j] += v.x * u.x + v.y * u.y + v.z * u.z + v.w * u.w;
            }
        }
        float mx = a[0];
        #pragma unroll
        for (int j = 1; j < OUTC; ++j) mx = fmaxf(mx, a[j]);
        float s = 0.0f;
        #pragma unroll
        for (int j = 0; j < OUTC; ++j) { a[j] = __expf(a[j] - mx); s += a[j]; }
        float rs = 1.0f / s;
        #pragma unroll
        for (int j = 0; j < OUTC; ++j) out[node * OUTC + j] = a[j] * rs;
    }
}

// ---------------------------------------------------------------------------
extern "C" void kernel_launch(void* const* d_in, const int* in_sizes, int n_in,
                              void* d_out, int out_size, void* d_ws, size_t ws_size,
                              hipStream_t stream) {
    const float* x_sub = (const float*)d_in[0];
    const float* x_agr = (const float*)d_in[1];
    const float* x_urb = (const float*)d_in[2];
    const float* Wl    = (const float*)d_in[3];
    const float* bl    = (const float*)d_in[4];
    const float* Wr    = (const float*)d_in[5];
    const float* Wf    = (const float*)d_in[6];
    const float* bf    = (const float*)d_in[7];
    const int*   e_ss  = (const int*)  d_in[8];
    const int*   as_s  = (const int*)  d_in[9];
    const int*   as_d  = (const int*)  d_in[10];
    const int*   e_us  = (const int*)  d_in[11];

    const int N_SUB = in_sizes[0] / H;
    const int E_SS  = in_sizes[8] / 2;
    const int E_AS  = in_sizes[9];
    const int E_US  = in_sizes[11] / 2;
    const int NRB   = N_SUB / 16;                 // 12500
    const int NF    = N_SUB * H;                  // floats per node matrix

    const int* ss_src = e_ss;           const int* ss_dst = e_ss + E_SS;
    const int* us_src = e_us;           const int* us_dst = e_us + E_US;

    // workspace layout (floats)
    float* agg_as = (float*)d_ws;
    float* agg_us = agg_as + NF;
    float* agg_ss = agg_us + NF;
    float* h0     = agg_ss + NF;
    float* h1     = h0     + NF;
    float* degv   = h1     + NF;
    uint32_t* Bswz = (uint32_t*)(degv + N_SUB);   // 3*8192 dwords
    float* bsum   = (float*)(Bswz + 3 * 8192);    // 3*64

    const int T = 256;
    // ---- layer-invariant aggregations (once per call) ----
    k_zero<<<dim3(2048), dim3(T), 0, stream>>>(agg_as, 2 * NF);        // agg_as+agg_us
    k_zero<<<dim3(64),   dim3(T), 0, stream>>>(degv, N_SUB);
    k_scatter<<<dim3((E_AS * 16 + T - 1) / T), dim3(T), 0, stream>>>(
        (const float4*)x_agr, as_s, as_d, agg_as, E_AS);
    k_scatter<<<dim3((E_US * 16 + T - 1) / T), dim3(T), 0, stream>>>(
        (const float4*)x_urb, us_src, us_dst, agg_us, E_US);
    k_deg<<<dim3((E_SS + T - 1) / T), dim3(T), 0, stream>>>(ss_dst, degv, E_SS);

    // ---- weight prep: fragment-swizzled f16 B + summed bias ----
    k_prep_b<<<dim3((24576 + T - 1) / T), dim3(T), 0, stream>>>(Wl, Wr, Bswz);
    k_prep_bias<<<dim3(1), dim3(T), 0, stream>>>(bl, bsum);

    // ---- 3 SAGE layers ----
    const float* cur = x_sub;
    float* bufs[2] = { h0, h1 };
    for (int layer = 0; layer < 3; ++layer) {
        k_zero<<<dim3(1024), dim3(T), 0, stream>>>(agg_ss, NF);
        k_scatter<<<dim3((E_SS * 16 + T - 1) / T), dim3(T), 0, stream>>>(
            (const float4*)cur, ss_src, ss_dst, agg_ss, E_SS);
        float* nxt = bufs[layer & 1];
        k_gemm<<<dim3(2048), dim3(GEMM_THREADS), 0, stream>>>(
            agg_ss, agg_as, agg_us, cur, degv,
            Bswz + layer * 8192, bsum + layer * H,
            nxt, NRB, layer > 0 ? 1 : 0);
        cur = nxt;
    }

    // ---- classifier + softmax ----
    k_head<<<dim3(1024), dim3(T), 0, stream>>>(cur, Wf, bf, (float*)d_out, N_SUB);
}
